// DeepLSTM_54932631716302
// MI455X (gfx1250) — compile-verified, matrix operand order checked by reading, not run on hardware
//
#include <hip/hip_runtime.h>

// ---------------------------------------------------------------------------
// DeepLSTM for MI455X (gfx1250, wave32, WMMA).
//
// Phase 0: transpose/convert all 8 MLP weight sets to bf16 Wt[N][K] in ws
//          (WMMA B-fragments load like A-fragments of W^T), copy biases,
//          zero sync counters.
// Phase 1: fully parallel precompute of Sx[g][b*T+t][128] = MLP_x_g(x[b,t])
//          (bf16 WMMA, fp32 accum). 8192 workgroups.
// Phase 2: 64 persistent workgroups = 16 row-tile groups x 4 gates.
//          Each WG caches its gate's h-MLP weights (262KB bf16) in LDS via
//          global_load_async_to_lds_b128, then runs the 512-step scan on its
//          16 batch rows; gates exchange fp32 pre-activations through L2 with
//          an atomic arrive counter (double-buffered by step parity).
// ---------------------------------------------------------------------------

typedef __attribute__((ext_vector_type(16))) __bf16 v16bf;
typedef __attribute__((ext_vector_type(8)))  float  v8f;

#define HID   128
#define T_SEQ 512
#define BATCH 256
#define BT    (BATCH * T_SEQ)

// per-MLP bf16 transposed-weight elements: 256x128 + 256x256 + 128x256
constexpr size_t WT_PER_MLP = 131072;
constexpr size_t OFF_WT   = 0;                          // 8 * 262144 B
constexpr size_t OFF_BIAS = 8 * WT_PER_MLP * 2;         // 2,097,152
constexpr size_t OFF_ARR  = OFF_BIAS + 8 * 640 * 4;     // 16 ints
constexpr size_t OFF_SBUF = (OFF_ARR + 64 + 255) & ~size_t(255);
// Sbuf: [16 groups][2 parity][4 gates][16][128] fp32 = 1 MB
constexpr size_t OFF_SX   = size_t(4) << 20;            // 4 * BT * 128 bf16 = 134 MB

constexpr int P1_LDS = (64 * 128 + 2 * 64 * 256) * 2;             // 81,920 B
constexpr int P2_LDS = 262144 + (16*128 + 2*16*256) * 2 + 16*128*4; // 290,816 B

// ---- WMMA helpers ---------------------------------------------------------

static __device__ __forceinline__ v16bf load_frag(const __bf16* base, int row,
                                                  int stride, int k0, int lane) {
  // lane l holds matrix row (row + l%16), 16 consecutive K values at
  // k0 + (l/16)*16.  Works for A (M-rows) and B (= A-layout of W^T).
  int r = lane & 15, h = lane >> 4;
  return *(const v16bf*)(base + (size_t)(row + r) * stride + k0 + h * 16);
}

static __device__ __forceinline__ v8f wmma_bf16(v16bf a, v16bf b, v8f c) {
  return __builtin_amdgcn_wmma_f32_16x16x32_bf16(false, a, false, b,
                                                 (short)0, c, false, false);
}

// one 16x16 output tile: rows [m0,m0+16) of A (MxK) times cols [n0,n0+16)
// of W (KxN) given as Wt (NxK, row-major, bf16)
static __device__ __forceinline__ v8f tile_mm(const __bf16* A, int astride, int m0,
                                              const __bf16* Wt, int wstride, int n0,
                                              int K, int lane) {
  v8f acc = {};
  for (int k = 0; k < K; k += 32) {
    v16bf a = load_frag(A,  m0, astride, k, lane);
    v16bf b = load_frag(Wt, n0, wstride, k, lane);
    acc = wmma_bf16(a, b, acc);
  }
  return acc;
}

// C layout: vgpr r, lanes0-15 -> M=r, lanes16-31 -> M=8+r; N = lane%16
static __device__ __forceinline__ void store_act(v8f acc, const float* bias, int n0,
                                                 __bf16* out, int ostride, int m0,
                                                 bool relu, int lane) {
  int n = lane & 15, h = lane >> 4;
  float bv = bias[n0 + n];
#pragma unroll
  for (int r = 0; r < 8; ++r) {
    float v = acc[r] + bv;
    if (relu) v = fmaxf(v, 0.f);
    out[(size_t)(m0 + r + 8 * h) * ostride + n0 + n] = (__bf16)v;
  }
}

// ---- Phase 0: weight prep -------------------------------------------------

struct MLPPtrs { const float *W0, *b0, *W1, *b1, *W2, *b2; };
struct AllParams { MLPPtrs m[8]; };

__global__ void __launch_bounds__(256) phase0_prep(AllParams P, char* __restrict__ ws) {
  int tid = blockIdx.x * 256 + threadIdx.x;
  if (blockIdx.x == 0 && threadIdx.x < 16)
    ((int*)(ws + OFF_ARR))[threadIdx.x] = 0;

  __bf16* WT = (__bf16*)(ws + OFF_WT);
  float*  BS = (float*)(ws + OFF_BIAS);
  const int total = 8 * 131072;
  for (int e = tid; e < total; e += gridDim.x * 256) {
    int m = e >> 17;
    int r = e & 131071;
    const MLPPtrs& p = P.m[m];
    float v;
    if (r < 32768)       { int n = r >> 7, k = r & 127;                 v = p.W0[k * 256 + n]; }
    else if (r < 98304)  { int r2 = r - 32768; int n = r2 >> 8, k = r2 & 255; v = p.W1[k * 256 + n]; }
    else                 { int r3 = r - 98304; int n = r3 >> 8, k = r3 & 255; v = p.W2[k * 128 + n]; }
    WT[e] = (__bf16)v;
  }
  for (int e = tid; e < 8 * 640; e += gridDim.x * 256) {
    int m = e / 640, r = e % 640;
    const MLPPtrs& p = P.m[m];
    BS[e] = (r < 256) ? p.b0[r] : (r < 512 ? p.b1[r - 256] : p.b2[r - 512]);
  }
}

// ---- Phase 1: x-MLP precompute (parallel over all B*T samples) ------------

__global__ void __launch_bounds__(256) phase1_xmlp(const float* __restrict__ x,
                                                   char* __restrict__ ws) {
  extern __shared__ __align__(32) char smem1[];
  __bf16* As  = (__bf16*)smem1;        // 64 x 128
  __bf16* Ab1 = As  + 64 * 128;        // 64 x 256
  __bf16* Ab2 = Ab1 + 64 * 256;        // 64 x 256

  const int tile = blockIdx.x;         // 0..2047 (64 samples each)
  const int g    = blockIdx.y;         // gate 0..3
  const int mlp  = 4 + g;              // x-MLPs are params 4..7
  const __bf16* Wt = (const __bf16*)(ws + OFF_WT) + (size_t)mlp * WT_PER_MLP;
  const float*  B  = (const float*)(ws + OFF_BIAS) + mlp * 640;
  __bf16* Sx = (__bf16*)(ws + OFF_SX) + (size_t)g * BT * 128;

  const int tid = threadIdx.x, lane = tid & 31, wave = tid >> 5;
  const size_t s0 = (size_t)tile * 64; // x is [B,T,128] => sample-major

  for (int i = tid; i < 64 * 128; i += 256)
    As[i] = (__bf16)x[s0 * 128 + i];
  __syncthreads();

  // L1: 128 -> 256 (4 m-tiles x 16 n-tiles, 8 per wave)
  for (int t = wave * 8; t < wave * 8 + 8; ++t) {
    int m = t >> 4, n = t & 15;
    v8f acc = tile_mm(As, 128, m * 16, Wt, 128, n * 16, 128, lane);
    store_act(acc, B, n * 16, Ab1, 256, m * 16, true, lane);
  }
  __syncthreads();
  // L2: 256 -> 256
  for (int t = wave * 8; t < wave * 8 + 8; ++t) {
    int m = t >> 4, n = t & 15;
    v8f acc = tile_mm(Ab1, 256, m * 16, Wt + 32768, 256, n * 16, 256, lane);
    store_act(acc, B + 256, n * 16, Ab2, 256, m * 16, true, lane);
  }
  __syncthreads();
  // L3: 256 -> 128 (linear), write bf16 straight to Sx
  for (int t = wave * 4; t < wave * 4 + 4; ++t) {
    int m = t >> 3, n = t & 7;
    v8f acc = tile_mm(Ab2, 256, m * 16, Wt + 98304, 256, n * 16, 256, lane);
    int nn = lane & 15, h = lane >> 4;
    float bv = B[512 + n * 16 + nn];
#pragma unroll
    for (int r = 0; r < 8; ++r) {
      size_t row = s0 + m * 16 + r + 8 * h;
      Sx[row * 128 + n * 16 + nn] = (__bf16)(acc[r] + bv);
    }
  }
}

// ---- Phase 2: recurrent scan (64 persistent WGs) --------------------------

__global__ void __launch_bounds__(256) phase2_scan(char* __restrict__ ws,
                                                   float* __restrict__ out) {
  extern __shared__ __align__(32) char smem2[];
  __bf16* Wl = (__bf16*)smem2;                    // 131072 bf16 = 262144 B
  __bf16* Hn = (__bf16*)(smem2 + 262144);         // 16 x 128
  __bf16* A1 = Hn + 16 * 128;                     // 16 x 256
  __bf16* A2 = A1 + 16 * 256;                     // 16 x 256
  float*  Cn = (float*)(A2 + 16 * 256);           // 16 x 128 fp32

  const int wg = blockIdx.x;
  const int rb = wg >> 2;    // row-tile group (16 batch rows)
  const int g  = wg & 3;     // gate: 0=F 1=I 2=O 3=C_hat
  const int tid = threadIdx.x, lane = tid & 31, wave = tid >> 5;

  const __bf16* WTg = (const __bf16*)(ws + OFF_WT) + (size_t)g * WT_PER_MLP;
  const float*  B   = (const float*)(ws + OFF_BIAS) + g * 640;
  const __bf16* Sx  = (const __bf16*)(ws + OFF_SX) + (size_t)g * BT * 128;
  float* Sbuf = (float*)(ws + OFF_SBUF);
  int*   arrive = (int*)(ws + OFF_ARR);

  // --- async-copy this gate's weights (262144 B) into LDS ---
  {
    const char* gsrc = (const char*)WTg;
    for (int it = 0; it < 64; ++it) {
      unsigned off = (unsigned)(it * 4096 + tid * 16);   // LDS byte offset == global offset
      asm volatile("global_load_async_to_lds_b128 %0, %1, %2"
                   :: "v"(off), "v"(off), "s"(gsrc) : "memory");
    }
    asm volatile("s_wait_asynccnt 0x0" ::: "memory");
  }
  for (int i = tid; i < 16 * 128; i += 256) { Hn[i] = (__bf16)0.f; Cn[i] = 0.f; }
  __syncthreads();

  const __bf16* W0 = Wl;            // 256 x 128
  const __bf16* W1 = Wl + 32768;    // 256 x 256
  const __bf16* W2 = Wl + 98304;    // 128 x 256

  for (int t = 0; t < T_SEQ; ++t) {
    // L1: hn(16x128) -> 16x256, 16 n-tiles, 2 per wave
    for (int n = wave * 2; n < wave * 2 + 2; ++n) {
      v8f acc = tile_mm(Hn, 128, 0, W0, 128, n * 16, 128, lane);
      store_act(acc, B, n * 16, A1, 256, 0, true, lane);
    }
    __syncthreads();
    // L2: 16x256 -> 16x256
    for (int n = wave * 2; n < wave * 2 + 2; ++n) {
      v8f acc = tile_mm(A1, 256, 0, W1, 256, n * 16, 256, lane);
      store_act(acc, B + 256, n * 16, A2, 256, 0, true, lane);
    }
    __syncthreads();
    // L3: 16x256 -> 16x128 (+bias +Sx), 8 n-tiles, 1 per wave -> Sbuf
    float* Sout = Sbuf + ((((size_t)rb * 2 + (t & 1)) * 4 + g) * 16 * 128);
    {
      int n = wave;
      v8f acc = tile_mm(A2, 256, 0, W2, 256, n * 16, 256, lane);
      int nn = lane & 15, h = lane >> 4;
      float bv = B[512 + n * 16 + nn];
#pragma unroll
      for (int r = 0; r < 8; ++r) {
        int m = r + 8 * h;
        size_t srow = ((size_t)(rb * 16 + m) * T_SEQ + t) * 128;
        Sout[m * 128 + n * 16 + nn] = acc[r] + bv + (float)Sx[srow + n * 16 + nn];
      }
    }
    __threadfence();      // make our S stores device-visible
    __syncthreads();
    if (tid == 0) {
      __hip_atomic_fetch_add(&arrive[rb], 1, __ATOMIC_ACQ_REL, __HIP_MEMORY_SCOPE_AGENT);
      const int target = 4 * (t + 1);
      while (__hip_atomic_load(&arrive[rb], __ATOMIC_ACQUIRE,
                               __HIP_MEMORY_SCOPE_AGENT) < target)
        __builtin_amdgcn_s_sleep(1);
    }
    __syncthreads();
    __builtin_amdgcn_fence(__ATOMIC_ACQUIRE, "agent");

    // elementwise combine (each gate WG redundantly keeps cn/hn local)
    const float* Sb = Sbuf + (((size_t)rb * 2 + (t & 1)) * 4) * 16 * 128;
    for (int i = tid; i < 16 * 128; i += 256) {
      float f  = 1.f / (1.f + __expf(-Sb[0 * 2048 + i]));
      float ii = 1.f / (1.f + __expf(-Sb[1 * 2048 + i]));
      float o  = 1.f / (1.f + __expf(-Sb[2 * 2048 + i]));
      float ch = tanhf(Sb[3 * 2048 + i]);
      float c  = f * Cn[i] + ii * ch;
      Cn[i] = c;
      float h = o * tanhf(c);
      Hn[i] = (__bf16)h;
      if (t == T_SEQ - 1 && g == 0)
        out[(size_t)rb * 16 * 128 + i] = h;   // fp32 final hidden state
    }
    __syncthreads();
  }
}

// ---- launcher -------------------------------------------------------------

extern "C" void kernel_launch(void* const* d_in, const int* in_sizes, int n_in,
                              void* d_out, int out_size, void* d_ws, size_t ws_size,
                              hipStream_t stream) {
  (void)in_sizes; (void)n_in; (void)out_size; (void)ws_size;
  // input order: x, then for each of F_h,I_h,O_h,C_hat_h,F_x,I_x,O_x,C_hat_x:
  // W0,b0,W1,b1,W2,b2
  AllParams P;
  for (int m = 0; m < 8; ++m) {
    void* const* q = d_in + 1 + m * 6;
    P.m[m].W0 = (const float*)q[0];
    P.m[m].b0 = (const float*)q[1];
    P.m[m].W1 = (const float*)q[2];
    P.m[m].b1 = (const float*)q[3];
    P.m[m].W2 = (const float*)q[4];
    P.m[m].b2 = (const float*)q[5];
  }
  const float* x = (const float*)d_in[0];
  char* ws = (char*)d_ws;

  hipFuncSetAttribute((const void*)phase1_xmlp,
                      hipFuncAttributeMaxDynamicSharedMemorySize, P1_LDS);
  hipFuncSetAttribute((const void*)phase2_scan,
                      hipFuncAttributeMaxDynamicSharedMemorySize, P2_LDS);

  phase0_prep<<<1024, 256, 0, stream>>>(P, ws);
  dim3 g1(2048, 4);
  phase1_xmlp<<<g1, 256, P1_LDS, stream>>>(x, ws);
  phase2_scan<<<64, 256, P2_LDS, stream>>>(ws, (float*)d_out);
}
